// PointCNN_3839700762642
// MI455X (gfx1250) — matrix operand order repaired and checked by reading, not compiled, not run on hardware
//
#include <hip/hip_runtime.h>
#include <hip/hip_bf16.h>

// ---------------------------------------------------------------------------
// PointCNN X-conv layer for MI455X (gfx1250, wave32, WMMA f16->f32)
// N=32, NUM_PTS=2048, P=1024, K=16, DIMS=3, C_IN=64, C_OUT=128
// B weight panels pre-swizzled into WMMA fragment order so every B fragment
// is one contiguous 32B LDS load; A fragments are two contiguous 16B loads.
// ---------------------------------------------------------------------------

#define NB      32
#define NUMPTS  2048
#define PP      1024
#define KK      16
#define C_CAT   96
#define KSQ     256          // K*K
#define NPTOT   (NB * PP)    // 32768 rep points

static __device__ __constant__ float kBnInv = 0.9999950000374997f; // 1/sqrt(1+1e-5)

typedef __attribute__((ext_vector_type(16))) _Float16 v16h;
typedef __attribute__((ext_vector_type(8)))  _Float16 v8h;
typedef __attribute__((ext_vector_type(8)))  float    v8f;

__device__ inline v8f wmma16(v16h a, v16h b, v8f c) {
  // D = A(16x32 f16) * B(32x16 f16) + C(16x16 f32)
  return __builtin_amdgcn_wmma_f32_16x16x32_f16(false, a, false, b, (short)0, c,
                                                false, false);
}

__device__ inline float elu(float x) { return x > 0.f ? x : (__expf(x) - 1.f); }

// Swizzled B index: fragment order [k>>5][n>>4][lane][(k&7)+8*((k>>4)&1)] where
// lane = (n&15) + 16*((k>>3)&1).  One fragment = 16 contiguous halves.
__device__ inline int swB(int k, int n, int N) {
  return ((((k >> 5) * (N >> 4) + (n >> 4)) * 32 +
           ((n & 15) + (((k >> 3) & 1) << 4))) << 4) + ((k & 7) + (((k >> 4) & 1) << 3));
}

// A-fragment (16x32 tile, f16 row-major source). Elements 0..7 are k=kh..kh+7,
// elements 8..15 are k=16+kh..23+kh -> two contiguous 16B loads.
__device__ inline v16h frag_a_f16(const _Float16* A, int lda, int lane) {
  const _Float16* p = A + (lane & 15) * lda + ((lane >> 4) << 3);
  v8h lo = *reinterpret_cast<const v8h*>(p);
  v8h hi = *reinterpret_cast<const v8h*>(p + 16);
  return __builtin_shufflevector(lo, hi, 0, 1, 2, 3, 4, 5, 6, 7,
                                 8, 9, 10, 11, 12, 13, 14, 15);
}

// Same but converting from f32 on the fly (compiler emits b128 + v_cvt_pk).
__device__ inline v16h frag_a_f32(const float* A, int lda, int lane) {
  const float* p = A + (lane & 15) * lda + ((lane >> 4) << 3);
  v16h a;
#pragma unroll
  for (int i = 0; i < 8; ++i) {
    a[i]     = (_Float16)p[i];
    a[i + 8] = (_Float16)p[16 + i];
  }
  return a;
}

// B-fragment from a pre-swizzled LDS chunk: one 32B contiguous load.
__device__ inline v16h frag_b_sw(const _Float16* sB, int ct, int lane) {
  return *reinterpret_cast<const v16h*>(sB + ((ct * 32 + lane) << 4));
}

// ---------------------------------------------------------------------------
// K0: weight precast to f16, written directly in swizzled fragment order
// ---------------------------------------------------------------------------
__global__ __launch_bounds__(256) void k_prep(
    const float* d0W, const float* xcW, const float* xd1W, const float* xd2W,
    const float* pwW, _Float16* Wd0, _Float16* Wxc, _Float16* Wxd1,
    _Float16* Wxd2, _Float16* Wpw) {
  int i = blockIdx.x * 256 + threadIdx.x;                 // 0..65535
  if (i < 64 * 64) {                                      // (cin,cout)
    int k = i >> 6, n = i & 63;
    Wd0[swB(k, n, 64)] = (_Float16)d0W[i];
  }
  if (i < 64 * KSQ) {                                     // padded 48->64 rows
    int k = i >> 8, n = i & 255;                          // k = d*16+kk
    Wxc[swB(k, n, 256)] = (k < 48) ? (_Float16)xcW[n * 48 + k] : (_Float16)0.f;
  }
  if (i < KSQ * KSQ) {
    int k = i >> 8, n = i & 255;
    int s = swB(k, n, 256);
    Wxd1[s] = (_Float16)xd1W[i];
    Wxd2[s] = (_Float16)xd2W[i];
  }
  if (i < 192 * 128) {
    int k = i >> 7, n = i & 127;
    Wpw[swB(k, n, 128)] = (_Float16)pwW[i];
  }
}

// ---------------------------------------------------------------------------
// K1: fts_l = BN(ELU(fts @ d0_W + b))   65536 x 64 x 64, out f16
// ---------------------------------------------------------------------------
__global__ __launch_bounds__(256) void k_lift(
    const float* fts, const _Float16* W, const float* b, const float* g,
    const float* be, _Float16* out) {
  __shared__ __align__(32) _Float16 sW[64 * 64];
  int t = threadIdx.x;
  for (int i = t; i < 1024; i += 256)
    reinterpret_cast<unsigned long long*>(sW)[i] =
        reinterpret_cast<const unsigned long long*>(W)[i];
  __syncthreads();

  int wave = t >> 5, lane = t & 31;
  int row0 = (blockIdx.x * 8 + wave) * 16;

  v8f z = {0.f, 0.f, 0.f, 0.f, 0.f, 0.f, 0.f, 0.f};
  v8f acc[4];
#pragma unroll
  for (int ct = 0; ct < 4; ++ct) acc[ct] = z;

#pragma unroll
  for (int kc = 0; kc < 2; ++kc) {
    v16h a = frag_a_f32(fts + (size_t)row0 * 64 + kc * 32, 64, lane);
#pragma unroll
    for (int ct = 0; ct < 4; ++ct)
      acc[ct] = wmma16(a, frag_b_sw(sW + kc * 2048, ct, lane), acc[ct]);
  }

  int nn = lane & 15, hh = lane >> 4;
#pragma unroll
  for (int ct = 0; ct < 4; ++ct) {
    int col = ct * 16 + nn;
    float bs = b[col], gg = g[col], bb = be[col];
#pragma unroll
    for (int r = 0; r < 8; ++r) {
      float x = elu(acc[ct][r] + bs);
      out[(size_t)(row0 + r + 8 * hh) * 64 + col] = (_Float16)(gg * (x * kBnInv) + bb);
    }
  }
}

// ---------------------------------------------------------------------------
// K2: neighbor gather + centering + tiny MLP (3->32->32), scalar per (n,p,k)
// ---------------------------------------------------------------------------
__global__ __launch_bounds__(256) void k_gather_mlp(
    const float* rep, const float* pts, const int* pts_idx,
    const float* d1W, const float* d1b, const float* d1g, const float* d1be,
    const float* d2W, const float* d2b, const float* d2g, const float* d2be,
    _Float16* ptsl, _Float16* hbuf) {
  __shared__ float s1W[96], s2W[1024];
  __shared__ float s1b[32], s1g[32], s1be[32], s2b[32], s2g[32], s2be[32];
  int t = threadIdx.x;
  for (int i = t; i < 96; i += 256) s1W[i] = d1W[i];
  for (int i = t; i < 1024; i += 256) s2W[i] = d2W[i];
  if (t < 32) {
    s1b[t] = d1b[t]; s1g[t] = d1g[t]; s1be[t] = d1be[t];
    s2b[t] = d2b[t]; s2g[t] = d2g[t]; s2be[t] = d2be[t];
  }
  __syncthreads();

  int g = blockIdx.x * 256 + t;                 // 0 .. NPTOT*KK-1
  int np = g >> 4, k = g & 15;
  int n = np >> 10;
  int idx = pts_idx[g];

  float lx = pts[((size_t)n * NUMPTS + idx) * 3 + 0] - rep[(size_t)np * 3 + 0];
  float ly = pts[((size_t)n * NUMPTS + idx) * 3 + 1] - rep[(size_t)np * 3 + 1];
  float lz = pts[((size_t)n * NUMPTS + idx) * 3 + 2] - rep[(size_t)np * 3 + 2];

  _Float16* pl = ptsl + (size_t)np * 64;
  pl[k]      = (_Float16)lx;
  pl[16 + k] = (_Float16)ly;
  pl[32 + k] = (_Float16)lz;
  pl[48 + k] = (_Float16)0.f;                   // zero pad K 48..63

  float h1[32];
#pragma unroll
  for (int j = 0; j < 32; ++j) {
    float v = lx * s1W[j] + ly * s1W[32 + j] + lz * s1W[64 + j] + s1b[j];
    h1[j] = s1g[j] * (elu(v) * kBnInv) + s1be[j];
  }
  _Float16* hr = hbuf + (size_t)np * (16 * 32) + k * 32;
#pragma unroll
  for (int j = 0; j < 32; ++j) {
    float v = s2b[j];
#pragma unroll
    for (int i = 0; i < 32; ++i) v += h1[i] * s2W[i * 32 + j];
    hr[j] = (_Float16)(s2g[j] * (elu(v) * kBnInv) + s2be[j]);
  }
}

// ---------------------------------------------------------------------------
// K3: WMMA GEMM, M x K x 256, f16 in/out, fused bias (+optional ELU)
// 8 waves/block, each wave 16 rows x 256 cols. Swizzled B chunk in LDS.
// ---------------------------------------------------------------------------
__global__ __launch_bounds__(256) void k_gemm(
    const _Float16* A, const _Float16* B, const float* bias, _Float16* out,
    int K, int do_elu) {
  __shared__ __align__(32) _Float16 sB[32 * 256];
  int t = threadIdx.x, wave = t >> 5, lane = t & 31;
  int row0 = (blockIdx.x * 8 + wave) * 16;

  v8f z = {0.f, 0.f, 0.f, 0.f, 0.f, 0.f, 0.f, 0.f};
  v8f acc[16];
#pragma unroll
  for (int ct = 0; ct < 16; ++ct) acc[ct] = z;

  int kchunks = K >> 5;
  for (int kc = 0; kc < kchunks; ++kc) {
    const _Float16* Bk = B + (size_t)kc * 32 * 256;   // chunk-contiguous
    if (kc + 1 < kchunks)
      __builtin_prefetch(B + (size_t)(kc + 1) * 32 * 256, 0, 1);
    for (int i = t; i < 2048; i += 256)
      reinterpret_cast<unsigned long long*>(sB)[i] =
          reinterpret_cast<const unsigned long long*>(Bk)[i];
    __syncthreads();

    v16h a = frag_a_f16(A + (size_t)row0 * K + kc * 32, K, lane);
#pragma unroll
    for (int ct = 0; ct < 16; ++ct)
      acc[ct] = wmma16(a, frag_b_sw(sB, ct, lane), acc[ct]);
    __syncthreads();
  }

  int nn = lane & 15, hh = lane >> 4;
#pragma unroll
  for (int ct = 0; ct < 16; ++ct) {
    int col = ct * 16 + nn;
    float bs = bias[col];
#pragma unroll
    for (int r = 0; r < 8; ++r) {
      float x = acc[ct][r] + bs;
      if (do_elu) x = elu(x);
      out[(size_t)(row0 + r + 8 * hh) * 256 + col] = (_Float16)x;
    }
  }
}

// ---------------------------------------------------------------------------
// K4: per rep point: fts_X = X(16x16) @ cat(16x96) (zero-padded K=32 WMMA),
// then depthwise (1,K) conv -> dwcat[np][192] f16.
// 128 threads = 4 waves, one point per wave. cat staged column-major so
// each B fragment is one 16B LDS load plus a zero half.
// ---------------------------------------------------------------------------
__global__ __launch_bounds__(128) void k_point(
    const _Float16* X2, const _Float16* hbuf, const _Float16* ftsl,
    const int* pts_idx, const float* dwW, const float* dwb, _Float16* dwcat) {
  __shared__ __align__(32) _Float16 scatT[4][C_CAT * 16]; // [c][k] per wave
  __shared__ float sfx[4][16 * C_CAT];                    // fts_X row-major
  __shared__ int   sidx[4][16];
  int t = threadIdx.x, wave = t >> 5, lane = t & 31;
  int np = blockIdx.x * 4 + wave;
  int n = np >> 10;

  if (lane < 16) sidx[wave][lane] = pts_idx[(size_t)np * 16 + lane];
  __syncthreads();

  // stage cat^T: 3 columns per lane; col c<32 from h, else gathered fts_l
#pragma unroll
  for (int q = 0; q < 3; ++q) {
    int c = lane * 3 + q;
    v16h col;
    if (c < 32) {
      const _Float16* hp = hbuf + (size_t)np * 512 + c;
#pragma unroll
      for (int k = 0; k < 16; ++k) col[k] = hp[k * 32];
    } else {
      const _Float16* fb = ftsl + (size_t)n * NUMPTS * 64 + (c - 32);
#pragma unroll
      for (int k = 0; k < 16; ++k) col[k] = fb[(size_t)sidx[wave][k] * 64];
    }
    *reinterpret_cast<v16h*>(&scatT[wave][c * 16]) = col;
  }
  __syncthreads();

  // A = X reshaped 16x16, zero padded to K=32: one 16B load + zero half
  const _Float16* xp = X2 + (size_t)np * 256 + (lane & 15) * 16 + ((lane >> 4) << 3);
  v8h lo = *reinterpret_cast<const v8h*>(xp);
  v8h z8 = {0, 0, 0, 0, 0, 0, 0, 0};
  v16h a = __builtin_shufflevector(lo, z8, 0, 1, 2, 3, 4, 5, 6, 7,
                                   8, 9, 10, 11, 12, 13, 14, 15);

  v16h z16;
#pragma unroll
  for (int i = 0; i < 16; ++i) z16[i] = (_Float16)0.f;
  v8f zf = {0.f, 0.f, 0.f, 0.f, 0.f, 0.f, 0.f, 0.f};
  v8f acc[6];
  int nc = lane & 15;
#pragma unroll
  for (int ct = 0; ct < 6; ++ct) {
    v8h bl = *reinterpret_cast<const v8h*>(&scatT[wave][(ct * 16 + nc) * 16 + ((lane >> 4) << 3)]);
    v16h b = __builtin_shufflevector(bl, z8, 0, 1, 2, 3, 4, 5, 6, 7,
                                     8, 9, 10, 11, 12, 13, 14, 15);
    acc[ct] = wmma16(a, b, zf);
  }

  int hh = lane >> 4;
#pragma unroll
  for (int ct = 0; ct < 6; ++ct)
#pragma unroll
    for (int r = 0; r < 8; ++r)
      sfx[wave][(r + 8 * hh) * C_CAT + ct * 16 + nc] = acc[ct][r];
  __syncthreads();

  // depthwise: dw[c,m] = sum_k fts_X[k,c] * dw_W[c,m,k] + dw_b ; order c*2+m
#pragma unroll
  for (int q = 0; q < 6; ++q) {
    int o = lane * 6 + q;                 // 0..191 == c*2+m
    int c = o >> 1;
    float s = dwb[o];
#pragma unroll
    for (int k = 0; k < 16; ++k) s += sfx[wave][k * C_CAT + c] * dwW[o * 16 + k];
    dwcat[(size_t)np * 192 + o] = (_Float16)s;
  }
}

// ---------------------------------------------------------------------------
// K5: pointwise conv + ELU + BN -> d_out (f32). 32768 x 192 x 128 WMMA GEMM.
// ---------------------------------------------------------------------------
__global__ __launch_bounds__(256) void k_pw(
    const _Float16* A, const _Float16* B, const float* bias, const float* g,
    const float* be, float* out) {
  __shared__ __align__(32) _Float16 sB[32 * 128];
  int t = threadIdx.x, wave = t >> 5, lane = t & 31;
  int row0 = (blockIdx.x * 8 + wave) * 16;

  v8f z = {0.f, 0.f, 0.f, 0.f, 0.f, 0.f, 0.f, 0.f};
  v8f acc[8];
#pragma unroll
  for (int ct = 0; ct < 8; ++ct) acc[ct] = z;

#pragma unroll
  for (int kc = 0; kc < 6; ++kc) {
    const _Float16* Bk = B + (size_t)kc * 32 * 128;
    if (kc + 1 < 6) __builtin_prefetch(B + (size_t)(kc + 1) * 32 * 128, 0, 1);
    for (int i = t; i < 1024; i += 256)
      reinterpret_cast<unsigned long long*>(sB)[i] =
          reinterpret_cast<const unsigned long long*>(Bk)[i];
    __syncthreads();

    v16h a = frag_a_f16(A + (size_t)row0 * 192 + kc * 32, 192, lane);
#pragma unroll
    for (int ct = 0; ct < 8; ++ct)
      acc[ct] = wmma16(a, frag_b_sw(sB, ct, lane), acc[ct]);
    __syncthreads();
  }

  int nn = lane & 15, hh = lane >> 4;
#pragma unroll
  for (int ct = 0; ct < 8; ++ct) {
    int col = ct * 16 + nn;
    float bs = bias[col], gg = g[col], bb = be[col];
#pragma unroll
    for (int r = 0; r < 8; ++r) {
      float e = elu(acc[ct][r] + bs);
      out[(size_t)(row0 + r + 8 * hh) * 128 + col] = gg * (e * kBnInv) + bb;
    }
  }
}

// ---------------------------------------------------------------------------
extern "C" void kernel_launch(void* const* d_in, const int* in_sizes, int n_in,
                              void* d_out, int out_size, void* d_ws,
                              size_t ws_size, hipStream_t stream) {
  const float* rep    = (const float*)d_in[0];
  const float* pts    = (const float*)d_in[1];
  const float* fts    = (const float*)d_in[2];
  const int*   idx    = (const int*)d_in[3];
  const float* d0W    = (const float*)d_in[4];
  const float* d0b    = (const float*)d_in[5];
  const float* d0g    = (const float*)d_in[6];
  const float* d0be   = (const float*)d_in[7];
  const float* d1W    = (const float*)d_in[8];
  const float* d1b    = (const float*)d_in[9];
  const float* d1g    = (const float*)d_in[10];
  const float* d1be   = (const float*)d_in[11];
  const float* d2W    = (const float*)d_in[12];
  const float* d2b    = (const float*)d_in[13];
  const float* d2g    = (const float*)d_in[14];
  const float* d2be   = (const float*)d_in[15];
  const float* xcW    = (const float*)d_in[16];
  const float* xcb    = (const float*)d_in[17];
  const float* xd1W   = (const float*)d_in[18];
  const float* xd1b   = (const float*)d_in[19];
  const float* xd2W   = (const float*)d_in[20];
  const float* xd2b   = (const float*)d_in[21];
  const float* dwW    = (const float*)d_in[22];
  const float* dwb    = (const float*)d_in[23];
  const float* pwW    = (const float*)d_in[24];
  const float* pwb    = (const float*)d_in[25];
  const float* endg   = (const float*)d_in[26];
  const float* endbe  = (const float*)d_in[27];
  float* out = (float*)d_out;

  char* ws = (char*)d_ws;
  size_t off = 0;
  auto take = [&](size_t bytes) -> char* {
    char* p = ws + off;
    off = (off + bytes + 255) & ~(size_t)255;
    return p;
  };
  _Float16* Wd0   = (_Float16*)take((size_t)64 * 64 * 2);
  _Float16* Wxc   = (_Float16*)take((size_t)64 * 256 * 2);
  _Float16* Wxd1  = (_Float16*)take((size_t)256 * 256 * 2);
  _Float16* Wxd2  = (_Float16*)take((size_t)256 * 256 * 2);
  _Float16* Wpw   = (_Float16*)take((size_t)192 * 128 * 2);
  _Float16* ftsl  = (_Float16*)take((size_t)NB * NUMPTS * 64 * 2);
  _Float16* ptsl  = (_Float16*)take((size_t)NPTOT * 64 * 2);
  _Float16* hbuf  = (_Float16*)take((size_t)NPTOT * 16 * 32 * 2);
  _Float16* X0    = (_Float16*)take((size_t)NPTOT * 256 * 2);
  _Float16* X1    = (_Float16*)take((size_t)NPTOT * 256 * 2);
  _Float16* X2    = (_Float16*)take((size_t)NPTOT * 256 * 2);
  _Float16* dwcat = (_Float16*)take((size_t)NPTOT * 192 * 2);

  k_prep<<<256, 256, 0, stream>>>(d0W, xcW, xd1W, xd2W, pwW,
                                  Wd0, Wxc, Wxd1, Wxd2, Wpw);
  k_lift<<<512, 256, 0, stream>>>(fts, Wd0, d0b, d0g, d0be, ftsl);
  k_gather_mlp<<<2048, 256, 0, stream>>>(rep, pts, idx, d1W, d1b, d1g, d1be,
                                         d2W, d2b, d2g, d2be, ptsl, hbuf);
  k_gemm<<<256, 256, 0, stream>>>(ptsl, Wxc, xcb, X0, 64, 1);
  k_gemm<<<256, 256, 0, stream>>>(X0, Wxd1, xd1b, X1, 256, 1);
  k_gemm<<<256, 256, 0, stream>>>(X1, Wxd2, xd2b, X2, 256, 0);
  k_point<<<8192, 128, 0, stream>>>(X2, hbuf, ftsl, idx, dwW, dwb, dwcat);
  k_pw<<<256, 256, 0, stream>>>(dwcat, Wpw, pwb, endg, endbe, out);
}
